// PyG_PointNet2_Alpha_Predictor_11467562680982
// MI455X (gfx1250) — compile-verified, matrix operand order checked by reading, not run on hardware
//
#include <hip/hip_runtime.h>
#include <math.h>

typedef __attribute__((ext_vector_type(16))) _Float16 v16h;
typedef __attribute__((ext_vector_type(8)))  _Float16 v8h;
typedef __attribute__((ext_vector_type(8)))  float    v8f;
typedef _Float16 half_t;

// ---------------------------------------------------------------------------
// Weight pre-pack: f32 [K, N] row-major  ->  f16 fragment-major, K zero-padded
// to Kpad (multiple of 64). Layout: [kt][nt][lane][16 halves contiguous],
// following the WMMA 16-bit B-fragment k-mapping:
//   e in 0..7  -> k = kt*32 + (lane/16)*8 + e
//   e in 8..15 -> k = kt*32 + 16 + (lane/16)*8 + (e-8)
//   n = nt*16 + (lane & 15)
// So each lane's B fragment in the GEMM is two contiguous b128 loads.
// ---------------------------------------------------------------------------
__global__ __launch_bounds__(256)
void pack_weight_kernel(const float* __restrict__ W, half_t* __restrict__ Wp,
                        int K, int Kpad, int N)
{
  const int t = blockIdx.x * blockDim.x + threadIdx.x;
  const int total = Kpad * N;
  if (t >= total) return;
  const int ktBlock = N * 32;          // halves per kt block = (N/16)*512
  const int kt   = t / ktBlock;
  const int r    = t % ktBlock;
  const int nt   = r / 512;
  const int q    = r % 512;
  const int lane = q / 16;
  const int e    = q % 16;
  const int n    = nt * 16 + (lane & 15);
  const int kh   = lane >> 4;
  const int kl   = (e < 8) ? (kh * 8 + e) : (16 + kh * 8 + (e - 8));
  const int k    = kt * 32 + kl;
  Wp[t] = (k < K) ? (half_t)W[(size_t)k * N + n] : (half_t)0.0f;
}

// ---------------------------------------------------------------------------
// WMMA GEMM:  D[M,Nout](f16) = act(A[M,Kpad](f16) * W + bias),  f32 accumulate.
// A is row-major f16 with lda = Kpad (multiple of 64, zero-padded by producer).
// Each wave computes a 16x32 output tile (two 16x16 N-tiles) so the A fragment
// is loaded once per two WMMAs. 2-deep k unroll -> 4 independent accumulator
// chains, 12 b128 loads : 4 wmma per iteration. No guards in the hot loop;
// tile-bound exit is wave-uniform so EXEC is all-1s at the WMMAs.
// ---------------------------------------------------------------------------
__global__ __launch_bounds__(128)
void gemm_wmma_kernel(const half_t* __restrict__ A, int lda,
                      const half_t* __restrict__ Wp,
                      const float* __restrict__ bias, half_t* __restrict__ D,
                      int M, int Kpad, int Nout, int relu)
{
  const int lane   = threadIdx.x & 31;
  const int wave   = threadIdx.x >> 5;
  const int tileN0 = blockIdx.x * 2;            // two consecutive 16-col tiles
  const int tileM  = blockIdx.y * 4 + wave;
  const int mBase  = tileM * 16;
  if (mBase >= M) return;                       // wave-uniform
  const int mRow   = mBase + (lane & 15);
  const int khalf  = lane >> 4;

  const half_t* Ap  = A + (size_t)mRow * lda + (khalf << 3);
  const half_t* Bp0 = Wp + ((size_t)tileN0 * 32 + lane) * 16;
  const half_t* Bp1 = Bp0 + 512;                // next nt block
  const size_t bStride = (size_t)(Nout >> 4) * 512;   // halves per kt block
  const int nkt = Kpad >> 5;                          // even (Kpad % 64 == 0)

  v8f acc00 = {}, acc01 = {}, acc10 = {}, acc11 = {};
  for (int kt = 0; kt < nkt; kt += 2) {
    union { v16h v; v8h h[2]; } a0, a1, b00, b01, b10, b11;
    a0.h[0]  = *(const v8h*)(Ap);
    a0.h[1]  = *(const v8h*)(Ap + 16);
    b00.h[0] = *(const v8h*)(Bp0);
    b00.h[1] = *(const v8h*)(Bp0 + 8);
    b01.h[0] = *(const v8h*)(Bp1);
    b01.h[1] = *(const v8h*)(Bp1 + 8);
    a1.h[0]  = *(const v8h*)(Ap + 32);
    a1.h[1]  = *(const v8h*)(Ap + 48);
    b10.h[0] = *(const v8h*)(Bp0 + bStride);
    b10.h[1] = *(const v8h*)(Bp0 + bStride + 8);
    b11.h[0] = *(const v8h*)(Bp1 + bStride);
    b11.h[1] = *(const v8h*)(Bp1 + bStride + 8);
    acc00 = __builtin_amdgcn_wmma_f32_16x16x32_f16(false, a0.v, false, b00.v,
                                                   (short)0, acc00, false, false);
    acc01 = __builtin_amdgcn_wmma_f32_16x16x32_f16(false, a0.v, false, b01.v,
                                                   (short)0, acc01, false, false);
    acc10 = __builtin_amdgcn_wmma_f32_16x16x32_f16(false, a1.v, false, b10.v,
                                                   (short)0, acc10, false, false);
    acc11 = __builtin_amdgcn_wmma_f32_16x16x32_f16(false, a1.v, false, b11.v,
                                                   (short)0, acc11, false, false);
    Ap  += 64;
    Bp0 += 2 * bStride;
    Bp1 += 2 * bStride;
  }

  const int nCol0 = tileN0 * 16 + (lane & 15);
  const int nCol1 = nCol0 + 16;
  const float bv0 = bias[nCol0];
  const float bv1 = bias[nCol1];
  half_t* Dp = D + (size_t)(mBase + (khalf << 3)) * Nout;
#pragma unroll
  for (int r = 0; r < 8; ++r) {
    float v0 = acc00[r] + acc10[r] + bv0;
    float v1 = acc01[r] + acc11[r] + bv1;
    if (relu) { v0 = fmaxf(v0, 0.0f); v1 = fmaxf(v1, 0.0f); }
    Dp[nCol0] = (half_t)v0;
    Dp[nCol1] = (half_t)v1;
    Dp += Nout;
  }
}

// ---------------------------------------------------------------------------
// Farthest point sampling: one block per batch, iterative argmax with LDS
// tree reduction (the serial chain is the algorithm). f32 positions.
// ---------------------------------------------------------------------------
__global__ __launch_bounds__(256)
void fps_kernel(const float* __restrict__ pos, int* __restrict__ out_idx,
                int Npts, int nSamp)
{
  __shared__ float mind[4096];
  __shared__ float rv[256];
  __shared__ int   ri[256];
  __shared__ float curp[3];
  const int b = blockIdx.x, tid = threadIdx.x;
  const float* p = pos + (size_t)b * Npts * 3;
  int* oi = out_idx + (size_t)b * nSamp;

  if (tid == 0) { curp[0] = p[0]; curp[1] = p[1]; curp[2] = p[2]; oi[0] = 0; }
  __syncthreads();
  for (int i = tid; i < Npts; i += 256) {
    float dx = p[i*3+0] - curp[0], dy = p[i*3+1] - curp[1], dz = p[i*3+2] - curp[2];
    mind[i] = dx*dx + dy*dy + dz*dz;
  }
  __syncthreads();

  for (int s = 1; s < nSamp; ++s) {
    float bv = -1.0f; int bi = 0;
    for (int i = tid; i < Npts; i += 256) {
      float d = mind[i];
      if (d > bv) { bv = d; bi = i; }
    }
    rv[tid] = bv; ri[tid] = bi;
    __syncthreads();
    for (int offd = 128; offd > 0; offd >>= 1) {
      if (tid < offd) {
        if (rv[tid + offd] > rv[tid]) { rv[tid] = rv[tid + offd]; ri[tid] = ri[tid + offd]; }
      }
      __syncthreads();
    }
    const int nxt = ri[0];
    if (tid == 0) {
      oi[s] = nxt;
      curp[0] = p[nxt*3+0]; curp[1] = p[nxt*3+1]; curp[2] = p[nxt*3+2];
    }
    __syncthreads();
    const float cx = curp[0], cy = curp[1], cz = curp[2];
    for (int i = tid; i < Npts; i += 256) {
      float dx = p[i*3+0] - cx, dy = p[i*3+1] - cy, dz = p[i*3+2] - cz;
      float d = dx*dx + dy*dy + dz*dz;
      if (d < mind[i]) mind[i] = d;
    }
    __syncthreads();
  }
}

// ---------------------------------------------------------------------------
// Row gathers
// ---------------------------------------------------------------------------
__global__ __launch_bounds__(256)
void gather_f32_kernel(const float* __restrict__ src, const int* __restrict__ idx,
                       float* __restrict__ dst, int Nsrc, int Ndst, int C)
{
  const int b = blockIdx.y;
  const int t = blockIdx.x * blockDim.x + threadIdx.x;
  if (t >= Ndst * C) return;
  const int i = t / C, c = t % C;
  const int j = idx[(size_t)b * Ndst + i];
  dst[((size_t)b * Ndst + i) * C + c] = src[((size_t)b * Nsrc + j) * C + c];
}

__global__ __launch_bounds__(256)
void gather_f16_kernel(const half_t* __restrict__ src, const int* __restrict__ idx,
                       half_t* __restrict__ dst, int Nsrc, int Ndst, int C)
{
  const int b = blockIdx.y;
  const int t = blockIdx.x * blockDim.x + threadIdx.x;
  if (t >= Ndst * C) return;
  const int i = t / C, c = t % C;
  const int j = idx[(size_t)b * Ndst + i];
  dst[((size_t)b * Ndst + i) * C + c] = src[((size_t)b * Nsrc + j) * C + c];
}

// ---------------------------------------------------------------------------
// Channel concat with zero pad to Cout (multiple of 64): builds padded f16
// GEMM inputs. Variant ff: both operands f16; variant fp: second operand f32.
// ---------------------------------------------------------------------------
__global__ __launch_bounds__(256)
void concat_ff_pad_kernel(const half_t* __restrict__ A, int Ca,
                          const half_t* __restrict__ Bsrc, int Cb,
                          half_t* __restrict__ dst, int rows, int Cout)
{
  const int t = blockIdx.x * blockDim.x + threadIdx.x;
  if (t >= rows * Cout) return;
  const int r = t / Cout, c = t % Cout;
  half_t v = (half_t)0.0f;
  if (c < Ca)            v = A[(size_t)r * Ca + c];
  else if (c < Ca + Cb)  v = Bsrc[(size_t)r * Cb + (c - Ca)];
  dst[t] = v;
}

__global__ __launch_bounds__(256)
void concat_fp_pad_kernel(const half_t* __restrict__ A, int Ca,
                          const float* __restrict__ Bsrc, int Cb,
                          half_t* __restrict__ dst, int rows, int Cout)
{
  const int t = blockIdx.x * blockDim.x + threadIdx.x;
  if (t >= rows * Cout) return;
  const int r = t / Cout, c = t % Cout;
  half_t v = (half_t)0.0f;
  if (c < Ca)            v = A[(size_t)r * Ca + c];
  else if (c < Ca + Cb)  v = (half_t)Bsrc[(size_t)r * Cb + (c - Ca)];
  dst[t] = v;
}

// ---------------------------------------------------------------------------
// Brute-force 3-NN over f32 positions; stores indices + normalized
// inverse-square-distance weights.
// ---------------------------------------------------------------------------
__global__ __launch_bounds__(256)
void knn3_kernel(const float* __restrict__ pos_tgt, const float* __restrict__ pos_src,
                 int Nt, int Ns, int* __restrict__ kidx, float* __restrict__ kw)
{
  const int b = blockIdx.y;
  const int t = blockIdx.x * blockDim.x + threadIdx.x;
  if (t >= Nt) return;
  const float* pt = pos_tgt + ((size_t)b * Nt + t) * 3;
  const float tx = pt[0], ty = pt[1], tz = pt[2];
  const float* ps = pos_src + (size_t)b * Ns * 3;
  float d0 = 3.4e38f, d1 = 3.4e38f, d2 = 3.4e38f;
  int   i0 = 0, i1 = 0, i2 = 0;
  for (int s = 0; s < Ns; ++s) {
    const float dx = tx - ps[s*3+0], dy = ty - ps[s*3+1], dz = tz - ps[s*3+2];
    const float d = dx*dx + dy*dy + dz*dz;
    if (d < d0)      { d2 = d1; i2 = i1; d1 = d0; i1 = i0; d0 = d; i0 = s; }
    else if (d < d1) { d2 = d1; i2 = i1; d1 = d;  i1 = s; }
    else if (d < d2) { d2 = d;  i2 = s; }
  }
  const float w0 = 1.0f / fmaxf(d0, 1e-16f);
  const float w1 = 1.0f / fmaxf(d1, 1e-16f);
  const float w2 = 1.0f / fmaxf(d2, 1e-16f);
  const float inv = 1.0f / (w0 + w1 + w2);
  const size_t o = ((size_t)b * Nt + t) * 3;
  kidx[o+0] = i0; kidx[o+1] = i1; kidx[o+2] = i2;
  kw[o+0] = w0 * inv; kw[o+1] = w1 * inv; kw[o+2] = w2 * inv;
}

// ---------------------------------------------------------------------------
// kNN interpolation over f16 features (f32 math).
// ---------------------------------------------------------------------------
__global__ __launch_bounds__(256)
void interp_f16_kernel(const half_t* __restrict__ feat, const int* __restrict__ kidx,
                       const float* __restrict__ kw, half_t* __restrict__ out,
                       int Nt, int Ns, int C)
{
  const int b = blockIdx.y;
  const int t = blockIdx.x * blockDim.x + threadIdx.x;
  if (t >= Nt * C) return;
  const int i = t / C, c = t % C;
  const size_t o = ((size_t)b * Nt + i) * 3;
  const half_t* f = feat + (size_t)b * Ns * C;
  const float v = kw[o+0] * (float)f[(size_t)kidx[o+0] * C + c] +
                  kw[o+1] * (float)f[(size_t)kidx[o+1] * C + c] +
                  kw[o+2] * (float)f[(size_t)kidx[o+2] * C + c];
  out[((size_t)b * Nt + i) * C + c] = (half_t)v;
}

// ---------------------------------------------------------------------------
// Final head layer (64 -> 1) + softplus + MIN_ALPHA.
// ---------------------------------------------------------------------------
__global__ __launch_bounds__(256)
void head_final_kernel(const half_t* __restrict__ h, const float* __restrict__ W2,
                       const float* __restrict__ b2, float* __restrict__ out, int Mrows)
{
  __shared__ float w[64];
  const int tid = threadIdx.x;
  if (tid < 64) w[tid] = W2[tid];
  __syncthreads();
  const int t = blockIdx.x * blockDim.x + tid;
  if (t >= Mrows) return;
  const half_t* hp = h + (size_t)t * 64;
  float acc = b2[0];
#pragma unroll 8
  for (int c = 0; c < 64; ++c) acc += (float)hp[c] * w[c];
  const float sp = (acc > 20.0f) ? acc : log1pf(expf(acc));
  out[t] = sp + 0.01f;
}

// ---------------------------------------------------------------------------
// Orchestration
// ---------------------------------------------------------------------------
extern "C" void kernel_launch(void* const* d_in, const int* in_sizes, int n_in,
                              void* d_out, int out_size, void* d_ws, size_t ws_size,
                              hipStream_t stream)
{
  (void)in_sizes; (void)n_in; (void)out_size; (void)ws_size;
  const int B = 8, N = 4096;
  const int n1 = N / 4, n2 = N / 16, n3 = N / 64;     // 1024, 256, 64
  const float* pos = (const float*)d_in[0];

  // ---- workspace bump allocator ----
  char* base = (char*)d_ws;
  size_t off = 0;
  auto alloc = [&](size_t bytes) -> void* {
    void* p = base + off;
    off += (bytes + 255) & ~(size_t)255;
    return p;
  };

  // ---- layer table: (W idx, b idx, K, Kpad(mult 64), N) in flatten order ----
  struct L { int wi, bi, K, Kpad, N; };
  const L layers[17] = {
    {1,2,3,64,64},       {3,4,64,64,64},      {5,6,64,64,128},        // sa1
    {7,8,131,192,128},   {9,10,128,128,128},  {11,12,128,128,256},    // sa2
    {13,14,259,320,256}, {15,16,256,256,512}, {17,18,512,512,1024},   // sa3
    {19,20,1280,1280,256},{21,22,256,256,256},                        // fp3
    {23,24,384,384,256}, {25,26,256,256,128},                         // fp2
    {27,28,131,192,128}, {29,30,128,128,128}, {31,32,128,128,128},    // fp1
    {33,34,128,128,64}                                                // head L0
  };

  half_t* hw[17];
  for (int i = 0; i < 17; ++i) {
    const int n = layers[i].Kpad * layers[i].N;
    hw[i] = (half_t*)alloc((size_t)n * sizeof(half_t));
    pack_weight_kernel<<<(n + 255) / 256, 256, 0, stream>>>(
        (const float*)d_in[layers[i].wi], hw[i], layers[i].K, layers[i].Kpad, layers[i].N);
  }
  auto gemm = [&](const half_t* A, int li, half_t* D, int M) {
    const int Kp = layers[li].Kpad, Nd = layers[li].N;
    dim3 grid((unsigned)(Nd / 32), (unsigned)((M + 63) / 64));
    gemm_wmma_kernel<<<grid, 128, 0, stream>>>(
        A, Kp, hw[li], (const float*)d_in[layers[li].bi], D, M, Kp, Nd, 1);
  };

  // ---- persistent buffers ----
  int*    i1  = (int*)alloc(sizeof(int) * B * n1);
  int*    i2  = (int*)alloc(sizeof(int) * B * n2);
  int*    i3  = (int*)alloc(sizeof(int) * B * n3);
  float*  l1  = (float*)alloc(sizeof(float) * B * n1 * 3);
  float*  l2  = (float*)alloc(sizeof(float) * B * n2 * 3);
  float*  l3  = (float*)alloc(sizeof(float) * B * n3 * 3);
  half_t* f1  = (half_t*)alloc(sizeof(half_t) * (size_t)B * n1 * 128);
  half_t* f2  = (half_t*)alloc(sizeof(half_t) * (size_t)B * n2 * 256);
  half_t* f3  = (half_t*)alloc(sizeof(half_t) * (size_t)B * n3 * 1024);
  half_t* g2  = (half_t*)alloc(sizeof(half_t) * (size_t)B * n2 * 256);
  half_t* g1  = (half_t*)alloc(sizeof(half_t) * (size_t)B * n1 * 128);
  int*    ki2 = (int*)alloc(sizeof(int) * B * n2 * 3);
  float*  kw2 = (float*)alloc(sizeof(float) * B * n2 * 3);
  int*    ki1 = (int*)alloc(sizeof(int) * B * n1 * 3);
  float*  kw1 = (float*)alloc(sizeof(float) * B * n1 * 3);
  int*    ki0 = (int*)alloc(sizeof(int) * (size_t)B * N * 3);
  float*  kw0 = (float*)alloc(sizeof(float) * (size_t)B * N * 3);

  const size_t arenaBytes = (size_t)B * N * 192 * sizeof(half_t); // 12.6 MB, max padded act
  half_t* BUF0 = (half_t*)alloc(arenaBytes);
  half_t* BUF1 = (half_t*)alloc(arenaBytes);
  half_t* BUF2 = (half_t*)alloc(arenaBytes);

  // ================= SA1 =================
  fps_kernel<<<B, 256, 0, stream>>>(pos, i1, N, n1);
  gather_f32_kernel<<<dim3((n1*3 + 255)/256, B), 256, 0, stream>>>(pos, i1, l1, N, n1, 3);
  concat_fp_pad_kernel<<<((B*n1*64) + 255)/256, 256, 0, stream>>>(
      (const half_t*)nullptr, 0, l1, 3, BUF1, B * n1, 64);
  gemm(BUF1, 0, BUF0, B * n1);
  gemm(BUF0, 1, BUF1, B * n1);
  gemm(BUF1, 2, f1,   B * n1);

  // ================= SA2 =================
  fps_kernel<<<B, 256, 0, stream>>>(l1, i2, n1, n2);
  gather_f32_kernel<<<dim3((n2*3 + 255)/256, B), 256, 0, stream>>>(l1, i2, l2, n1, n2, 3);
  gather_f16_kernel<<<dim3((n2*128 + 255)/256, B), 256, 0, stream>>>(f1, i2, BUF0, n1, n2, 128);
  concat_fp_pad_kernel<<<((B*n2*192) + 255)/256, 256, 0, stream>>>(BUF0, 128, l2, 3, BUF1, B * n2, 192);
  gemm(BUF1, 3, BUF0, B * n2);
  gemm(BUF0, 4, BUF2, B * n2);
  gemm(BUF2, 5, f2,   B * n2);

  // ================= SA3 =================
  fps_kernel<<<B, 256, 0, stream>>>(l2, i3, n2, n3);
  gather_f32_kernel<<<dim3((n3*3 + 255)/256, B), 256, 0, stream>>>(l2, i3, l3, n2, n3, 3);
  gather_f16_kernel<<<dim3((n3*256 + 255)/256, B), 256, 0, stream>>>(f2, i3, BUF0, n2, n3, 256);
  concat_fp_pad_kernel<<<((B*n3*320) + 255)/256, 256, 0, stream>>>(BUF0, 256, l3, 3, BUF1, B * n3, 320);
  gemm(BUF1, 6, BUF0, B * n3);
  gemm(BUF0, 7, BUF2, B * n3);
  gemm(BUF2, 8, f3,   B * n3);

  // ================= FP3 =================
  knn3_kernel<<<dim3((n2 + 255)/256, B), 256, 0, stream>>>(l2, l3, n2, n3, ki2, kw2);
  interp_f16_kernel<<<dim3((n2*1024 + 255)/256, B), 256, 0, stream>>>(f3, ki2, kw2, BUF0, n2, n3, 1024);
  concat_ff_pad_kernel<<<((B*n2*1280) + 255)/256, 256, 0, stream>>>(BUF0, 1024, f2, 256, BUF1, B * n2, 1280);
  gemm(BUF1, 9,  BUF0, B * n2);
  gemm(BUF0, 10, g2,   B * n2);

  // ================= FP2 =================
  knn3_kernel<<<dim3((n1 + 255)/256, B), 256, 0, stream>>>(l1, l2, n1, n2, ki1, kw1);
  interp_f16_kernel<<<dim3((n1*256 + 255)/256, B), 256, 0, stream>>>(g2, ki1, kw1, BUF0, n1, n2, 256);
  concat_ff_pad_kernel<<<((B*n1*384) + 255)/256, 256, 0, stream>>>(BUF0, 256, f1, 128, BUF1, B * n1, 384);
  gemm(BUF1, 11, BUF0, B * n1);
  gemm(BUF0, 12, g1,   B * n1);

  // ================= FP1 =================
  knn3_kernel<<<dim3((N + 255)/256, B), 256, 0, stream>>>(pos, l1, N, n1, ki0, kw0);
  interp_f16_kernel<<<dim3((N*128 + 255)/256, B), 256, 0, stream>>>(g1, ki0, kw0, BUF0, N, n1, 128);
  concat_fp_pad_kernel<<<((B*N*192) + 255)/256, 256, 0, stream>>>(BUF0, 128, pos, 3, BUF1, B * N, 192);
  gemm(BUF1, 13, BUF0, B * N);
  gemm(BUF0, 14, BUF2, B * N);
  gemm(BUF2, 15, BUF1, B * N);       // g0

  // ================= Head =================
  gemm(BUF1, 16, BUF0, B * N);       // h = relu(g0 @ W1 + b1), [B*N, 64] f16
  head_final_kernel<<<(B*N + 255)/256, 256, 0, stream>>>(
      BUF0, (const float*)d_in[35], (const float*)d_in[36], (float*)d_out, B * N);
}
